// QuadConvLayer_80032420594372
// MI455X (gfx1250) — matrix operand order repaired and checked by reading, not compiled, not run on hardware
//
#include <hip/hip_runtime.h>

// ---------------- problem constants (from reference) ----------------
#define C_IN   8
#define C_OUT  8
#define HD     16
#define NTOT   400                 // mesh nodes (20x20 Gauss-Legendre)
#define MTOT   400                 // output locations
#define BSZ    16
#define DECAYF 625.0f              // (N1D/4)^4 = (20/4)^4
#define INV_DECAYF (1.0f/625.0f)
#define INV2PI 0.15915494309189535f // fold into weights: v_sin_f32(y) = sin(2*pi*y)

typedef __attribute__((ext_vector_type(16))) _Float16 v16h;
typedef __attribute__((ext_vector_type(8)))  float    v8f;

union V16HU { v16h h; unsigned int u[8]; };

__device__ __forceinline__ unsigned int pk2(float a, float b) {
  // v_cvt_pk_rtz_f16_f32 -> two f16 packed in a dword
  return __builtin_bit_cast(unsigned int, __builtin_amdgcn_cvt_pkrtz(a, b));
}
__device__ __forceinline__ float swapx16_f(float x) {
  // ds_swizzle SWAPX16: xor=0x10, and=0x1f (group-of-32 mode)
  return __builtin_bit_cast(float,
      (unsigned int)__builtin_amdgcn_ds_swizzle(
          (int)__builtin_bit_cast(unsigned int, x), 0x401F));
}

// One fused kernel: grid.x = C_OUT * 25 m-tiles = 200 blocks, 256 threads (8 waves).
// Wave w owns input channel i == w and loops the 13 n-chunks of 32, keeping a
// persistent 16(b) x 16(m) f32 WMMA accumulator. The W2 stage runs as
//   d_even = [W2|0] x B,  d_odd = [0|W2] x B
// so the raw per-lane packed h1 IS the shared B operand (no lane shuffles).
// Cross-wave (= cross-i) reduction through LDS at the end.
__global__ __launch_bounds__(256)
void quadconv_fused_kernel(const float* __restrict__ feat,   // (16,8,400)
                           const float* __restrict__ W1,     // (64,16,2)
                           const float* __restrict__ W2,     // (64,16,16)
                           const float* __restrict__ W3,     // (64,1,16)
                           const float* __restrict__ olocs,  // (400,2)
                           const float* __restrict__ nodes,  // (400,2)
                           const float* __restrict__ wq,     // (400,)
                           float* __restrict__ out)          // (16,8,400)
{
  __shared__ float red[8 * 256];

  const int o   = blockIdx.x & 7;          // output channel
  const int mt  = blockIdx.x >> 3;         // m-tile (0..24)
  const int m0  = mt * 16;

  const int tid  = threadIdx.x;
  const int lane = tid & 31;
  const int w    = __builtin_amdgcn_readfirstlane(tid >> 5); // wave id == input channel i
  const int l16  = lane & 15;              // serves as m-column AND as batch row b
  const bool lo  = (lane < 16);
  const int k    = o * C_IN + w;           // MLP index (wave-uniform)

  // output location for this lane's m (loop invariant)
  const float2 ol = reinterpret_cast<const float2*>(olocs)[m0 + l16];
  const float ox = ol.x, oy = ol.y;

  // ---- W1[k] pre-scaled by 1/(2*pi) (uniform -> SGPRs), loaded once ----
  float w1x[16], w1y[16];
  {
    const float* W1k = W1 + k * 32;
    #pragma unroll
    for (int h = 0; h < 16; ++h) {
      w1x[h] = W1k[2*h]   * INV2PI;
      w1y[h] = W1k[2*h+1] * INV2PI;
    }
  }

  const int c0 = lo ? 0 : 8;

  // ---- W2[k]/(2*pi) as TWO WMMA A operands (built once):
  //   A_lo = [W2 | 0 ]  -> contracts B rows 0..15  (h1 of even-n points)
  //   A_hi = [ 0 | W2]  -> contracts B rows 16..31 (h1 of odd-n points)
  // k order used here (contiguous, same in B): lane l, row=l%16; lanes<16: V0-7=K0..15,
  //                             lanes>=16: V0-7=K16..31.
  V16HU Aw2lo, Aw2hi;
  {
    const float4* r = reinterpret_cast<const float4*>(W2 + k*256 + l16*16);
    const float4 a = r[0], b = r[1], c = r[2], e = r[3];
    unsigned int p[8];
    p[0] = pk2(a.x*INV2PI, a.y*INV2PI);
    p[1] = pk2(a.z*INV2PI, a.w*INV2PI);
    p[2] = pk2(b.x*INV2PI, b.y*INV2PI);
    p[3] = pk2(b.z*INV2PI, b.w*INV2PI);
    p[4] = pk2(c.x*INV2PI, c.y*INV2PI);
    p[5] = pk2(c.z*INV2PI, c.w*INV2PI);
    p[6] = pk2(e.x*INV2PI, e.y*INV2PI);
    p[7] = pk2(e.z*INV2PI, e.w*INV2PI);
    #pragma unroll
    for (int j = 0; j < 8; ++j) { Aw2lo.u[j] = lo ? p[j] : 0u; Aw2hi.u[j] = lo ? 0u : p[j]; }
  }

  // ---- W3[k] half-row for the D-layout dot product (once) ----
  float w3c[8];
  {
    const float4* r = reinterpret_cast<const float4*>(W3 + k*16 + c0);
    const float4 a = r[0], b = r[1];
    w3c[0]=a.x; w3c[1]=a.y; w3c[2]=a.z; w3c[3]=a.w;
    w3c[4]=b.x; w3c[5]=b.y; w3c[6]=b.z; w3c[7]=b.w;
  }

  const float* fi = feat + l16 * (C_IN * NTOT) + w * NTOT;  // row b = l16, channel i = w
  const int kb = lo ? 0 : 16;

  v8f acc = {0.f,0.f,0.f,0.f,0.f,0.f,0.f,0.f};   // D: rows=b, cols=m
  const v8f cz = {0.f,0.f,0.f,0.f,0.f,0.f,0.f,0.f};

  // A operand of the quadrature contraction: (w[n]*f[b,i,n]) 16x32 f16.
  // OOB K-rows pair with zeroed B rows, so only the load BASE is clamped
  // (garbage-but-finite A values multiply zero). Branchless, vectorized.
  auto load_Afw = [&](int n0) -> V16HU {
    V16HU A;
    const int b1s = n0 + kb;
    const int b1 = (b1s > NTOT - 16) ? (NTOT - 16) : b1s; // keep float4 in bounds
    const int b2 = b1 + 8;
    const float4 fa = *reinterpret_cast<const float4*>(fi + b1);
    const float4 fb = *reinterpret_cast<const float4*>(fi + b1 + 4);
    const float4 fc = *reinterpret_cast<const float4*>(fi + b2);
    const float4 fd = *reinterpret_cast<const float4*>(fi + b2 + 4);
    const float4 wa = *reinterpret_cast<const float4*>(wq + b1);
    const float4 wb = *reinterpret_cast<const float4*>(wq + b1 + 4);
    const float4 wc = *reinterpret_cast<const float4*>(wq + b2);
    const float4 wd = *reinterpret_cast<const float4*>(wq + b2 + 4);
    A.u[0] = pk2(fa.x*wa.x, fa.y*wa.y);
    A.u[1] = pk2(fa.z*wa.z, fa.w*wa.w);
    A.u[2] = pk2(fb.x*wb.x, fb.y*wb.y);
    A.u[3] = pk2(fb.z*wb.z, fb.w*wb.w);
    A.u[4] = pk2(fc.x*wc.x, fc.y*wc.y);
    A.u[5] = pk2(fc.z*wc.z, fc.w*wc.w);
    A.u[6] = pk2(fd.x*wd.x, fd.y*wd.y);
    A.u[7] = pk2(fd.z*wd.z, fd.w*wd.w);
    return A;
  };

  // One pass of 32 points: lane p evaluates point (m = p%16, n = n0+2pp+(p>=16)).
  // Returns packed f16 (kv*bump for even n, odd n) for column m, valid in ALL lanes.
  auto eval_pass = [&](int n) -> unsigned int {
    const float2 nd = reinterpret_cast<const float2*>(nodes)[n];
    const float zx = ox - nd.x, zy = oy - nd.y;

    // h1 = sin(W1 z): per-lane packed f16 is directly the shared B operand
    // (lanes 0..15 -> rows 0..15 / even-n points, lanes 16..31 -> rows 16..31).
    V16HU Bh;
    #pragma unroll
    for (int j = 0; j < 8; ++j) {
      const float s0 = __builtin_amdgcn_sinf(w1x[2*j]  *zx + w1y[2*j]  *zy);
      const float s1 = __builtin_amdgcn_sinf(w1x[2*j+1]*zx + w1y[2*j+1]*zy);
      Bh.u[j] = pk2(s0, s1);
    }

    v8f d0 = __builtin_amdgcn_wmma_f32_16x16x32_f16(false, Aw2lo.h, false, Bh.h,
                                                    (short)0, cz, false, false);
    v8f d1 = __builtin_amdgcn_wmma_f32_16x16x32_f16(false, Aw2hi.h, false, Bh.h,
                                                    (short)0, cz, false, false);

    // kv = W3 . sin(2*pi*d): D rows 0..7 in lanes<16, rows 8..15 in lanes>=16
    float p0 = w3c[0] * __builtin_amdgcn_sinf(d0[0]);
    float p1 = w3c[0] * __builtin_amdgcn_sinf(d1[0]);
    #pragma unroll
    for (int r = 1; r < 8; ++r) {
      p0 += w3c[r] * __builtin_amdgcn_sinf(d0[r]);
      p1 += w3c[r] * __builtin_amdgcn_sinf(d1[r]);
    }
    const float kv0 = p0 + swapx16_f(p0);   // full dot, all lanes (col m = lane%16)
    const float kv1 = p1 + swapx16_f(p1);

    // bump window (quadrature weight folded into the A operand)
    const float r2   = zx*zx + zy*zy;
    const float r4   = r2 * r2;
    const float safe = 1.f - DECAYF * r4;
    const float bump = (r4 < INV_DECAYF)
        ? __expf(1.f - __builtin_amdgcn_rcpf(safe)) : 0.f;

    // give every lane (bump_even, bump_odd) for its column m
    const float sb = swapx16_f(bump);
    const float be = lo ? bump : sb;
    const float bo = lo ? sb : bump;
    return pk2(kv0 * be, kv1 * bo);
  };

  auto accum_chunk = [&](const unsigned int* pkv, const V16HU& Afw) {
    // B operand (32 n x 16 m): pkv is valid in all lanes -> pure selects
    V16HU Bc;
    #pragma unroll
    for (int j = 0; j < 8; ++j) Bc.u[j] = lo ? pkv[j] : pkv[j + 8];
    acc = __builtin_amdgcn_wmma_f32_16x16x32_f16(false, Afw.h, false, Bc.h,
                                                 (short)0, acc, false, false);
  };

  // ---- 12 full chunks: no range guards anywhere ----
  #pragma unroll 1
  for (int ch = 0; ch < 12; ++ch) {
    const int n0 = ch * 32;
    const V16HU Afw = load_Afw(n0);
    unsigned int pkv[16];
    #pragma unroll
    for (int pp = 0; pp < 16; ++pp)
      pkv[pp] = eval_pass(n0 + 2*pp + (lane >> 4));
    accum_chunk(pkv, Afw);
  }

  // ---- tail chunk (n = 384..399): passes 8..15 are zero ----
  {
    const int n0 = 12 * 32;
    const V16HU Afw = load_Afw(n0);
    unsigned int pkv[16];
    #pragma unroll
    for (int pp = 0; pp < 8; ++pp)
      pkv[pp] = eval_pass(n0 + 2*pp + (lane >> 4));
    #pragma unroll
    for (int pp = 8; pp < 16; ++pp) pkv[pp] = 0u;
    accum_chunk(pkv, Afw);
  }

  // ---- cross-wave (cross-i) reduction of the 8 per-wave (b x m) accumulators ----
  #pragma unroll
  for (int r = 0; r < 8; ++r) red[w * 256 + r * 32 + lane] = acc[r];
  __syncthreads();

  const int b  = tid >> 4;
  const int m  = tid & 15;
  const int vg = b & 7;
  const int ln = (b < 8) ? m : (m + 16);     // D layout: M=vg(+8) at lanes 0-15/16-31
  float s = 0.f;
  #pragma unroll
  for (int ww = 0; ww < 8; ++ww) s += red[ww * 256 + vg * 32 + ln];
  out[b * (C_OUT * MTOT) + o * MTOT + (m0 + m)] = s;
}

extern "C" void kernel_launch(void* const* d_in, const int* in_sizes, int n_in,
                              void* d_out, int out_size, void* d_ws, size_t ws_size,
                              hipStream_t stream) {
  (void)in_sizes; (void)n_in; (void)out_size; (void)d_ws; (void)ws_size;
  const float* feat  = (const float*)d_in[0];
  const float* W1    = (const float*)d_in[1];
  const float* W2    = (const float*)d_in[2];
  const float* W3    = (const float*)d_in[3];
  const float* olocs = (const float*)d_in[4];
  const float* nodes = (const float*)d_in[5];
  const float* wq    = (const float*)d_in[6];
  float* out = (float*)d_out;

  dim3 grid(C_OUT * (MTOT / 16));   // 8 * 25 = 200 blocks
  dim3 block(256);                  // 8 wave32
  quadconv_fused_kernel<<<grid, block, 0, stream>>>(feat, W1, W2, W3,
                                                    olocs, nodes, wq, out);
}
